// MMD_13271448944952
// MI455X (gfx1250) — matrix operand order repaired and checked by reading, not compiled, not run on hardware
//
#include <hip/hip_runtime.h>
#include <hip/hip_bf16.h>

typedef __attribute__((ext_vector_type(16))) __bf16 v16bf;
typedef __attribute__((ext_vector_type(8)))  float  v8f;

#define B_ROWS    8192
#define HALF_ROWS 4096
#define DDIM      512
#define CDIM      5994
#define NTILES    47      // ceil(5994/128)
#define NCHUNKS   4
#define TPC       12      // class tiles per chunk
#define ASTRIDE   40      // bf16 row stride: 16B-aligned 8-elem runs, conflict-free
#define BSTRIDE   40

__device__ __forceinline__ void bsplit(float x, __bf16& hi, __bf16& lo) {
  hi = (__bf16)x;
  lo = (__bf16)(x - (float)hi);
}

// convert float4 -> hi/lo bf16 quads, stored as single b64 per plane
__device__ __forceinline__ void stage4(__bf16* hp, __bf16* lp, float4 v) {
  union { __bf16 b[4]; uint2 u; } ph, pl;
  bsplit(v.x, ph.b[0], pl.b[0]);
  bsplit(v.y, ph.b[1], pl.b[1]);
  bsplit(v.z, ph.b[2], pl.b[2]);
  bsplit(v.w, ph.b[3], pl.b[3]);
  *(uint2*)hp = ph.u;   // 8B-aligned: kp%4==0, row stride 80B
  *(uint2*)lp = pl.u;
}

// bf16x3 fp32-emulating accumulate: C += A*B with A=ahi+alo, B=bhi+blo
__device__ __forceinline__ v8f wmma3(v16bf ahi, v16bf alo, v16bf bhi, v16bf blo, v8f c) {
  c = __builtin_amdgcn_wmma_f32_16x16x32_bf16(false, ahi, false, bhi, (short)0, c, false, false);
  c = __builtin_amdgcn_wmma_f32_16x16x32_bf16(false, alo, false, bhi, (short)0, c, false, false);
  c = __builtin_amdgcn_wmma_f32_16x16x32_bf16(false, ahi, false, blo, (short)0, c, false, false);
  return c;
}

// A-operand 16-bit 16x32 (ISA 7.12.2): lane-half h=0 needs K 0..7,16..23; h=1: 8..15,24..31
__device__ __forceinline__ v16bf load_a16(const __bf16* rowp, int half) {
  v16bf v;
#pragma unroll
  for (int e = 0; e < 8; ++e) v[e] = rowp[half * 8 + e];
#pragma unroll
  for (int e = 0; e < 8; ++e) v[8 + e] = rowp[16 + half * 8 + e];
  return v;
}
// B-operand 32x16: same K order as A (h=0: 0..7,16..23; h=1: 8..15,24..31)
__device__ __forceinline__ v16bf load_b16(const __bf16* rowp, int half) {
  v16bf v;
#pragma unroll
  for (int e = 0; e < 8; ++e) v[e] = rowp[half * 8 + e];
#pragma unroll
  for (int e = 0; e < 8; ++e) v[8 + e] = rowp[16 + half * 8 + e];
  return v;
}

// sum over sigmas {1,2,5,10,20,40,80} of exp(-q/(2*sig)); powers of u=exp(-q/160)
__device__ __forceinline__ float expsum7(float q) {
  float u   = __expf(-0.00625f * q);  // sigma 80
  float u2  = u * u;                   // 40
  float u4  = u2 * u2;                 // 20
  float u8  = u4 * u4;                 // 10
  float u16 = u8 * u8;                 // 5
  float u32 = u16 * u16;
  float u40 = u32 * u8;                // 2
  float u80 = u40 * u40;               // 1
  return u + u2 + u4 + u8 + u16 + u40 + u80;
}

// ---------------- index scatter (faithful to reference _split_domains) ---------
__global__ void k_init_idx(int* main_src, int* targ_src) {
  int i = blockIdx.x * blockDim.x + threadIdx.x;
  if (i < HALF_ROWS) { main_src[i] = -1; targ_src[i] = -1; }
}

__global__ void k_scatter(const int* __restrict__ y_d, int* main_src, int* targ_src) {
  int i = blockIdx.x * blockDim.x + threadIdx.x;
  if (i >= B_ROWS) return;
  if (y_d[i] == 0) {
    if (i < HALF_ROWS) main_src[i] = i;
  } else {
    targ_src[i % HALF_ROWS] = i;
  }
}

// ---------------- row squared norms (Gram diagonals) ---------------------------
__global__ __launch_bounds__(128) void k_norms(const float* __restrict__ emb,
                                               const int* __restrict__ main_src,
                                               const int* __restrict__ targ_src,
                                               float* normM, float* normT) {
  __shared__ float red[128];
  const int row = blockIdx.x;
  const int src = (row < HALF_ROWS) ? main_src[row] : targ_src[row - HALF_ROWS];
  float s = 0.f;
  if (src >= 0) {
    for (int k = threadIdx.x; k < DDIM; k += 128) {
      float v = emb[src * DDIM + k];
      s += v * v;
    }
  }
  red[threadIdx.x] = s;
  __syncthreads();
  for (int off = 64; off > 0; off >>= 1) {
    if (threadIdx.x < off) red[threadIdx.x] += red[threadIdx.x + off];
    __syncthreads();
  }
  if (threadIdx.x == 0) {
    if (row < HALF_ROWS) normM[row] = red[0];
    else                 normT[row - HALF_ROWS] = red[0];
  }
}

// ---------------- classifier head: fused GEMM + streaming softmax --------------
// grid = (NCHUNKS, 8192/64); block = 256 (8 wave32). Block tile 64(M) x 128(N).
__global__ __launch_bounds__(256) void k_ce(const float* __restrict__ emb,
                                            const float* __restrict__ W,
                                            const int* __restrict__ y,
                                            float* pm, float* ps, float* pbest,
                                            float* pylog, int* pbidx) {
  __shared__ __bf16 Ah[64][ASTRIDE], Al[64][ASTRIDE];      // 10,240 B
  __shared__ __bf16 Bh[128][BSTRIDE], Bl[128][BSTRIDE];    // 20,480 B (W^T tile)
  __shared__ float  Lt[64][128];                           // 32,768 B
  const int tid = threadIdx.x;
  const int chunk = blockIdx.x;
  const int rowBase = blockIdx.y * 64;
  const int tile0 = chunk * TPC;
  const int tile1 = (tile0 + TPC < NTILES) ? (tile0 + TPC) : NTILES;
  const int lane = tid & 31, wv = tid >> 5;
  const int wm = wv & 1, wn = wv >> 1;
  const int half = lane >> 4, ln = lane & 15;

  float rm = -3.0e38f, rsum = 0.f, rbest = -3.0e38f, rylog = -3.0e38f;
  int rbidx = 0, yv = -1;
  if (tid < 128) yv = y[rowBase + (tid >> 1)];

  for (int nt = tile0; nt < tile1; ++nt) {
    const int n0 = nt * 128;
    v8f z = {};
    v8f acc00 = z, acc01 = z, acc10 = z, acc11 = z;

    for (int kk = 0; kk < DDIM; kk += 32) {
      // ---- load phase (registers only; overlaps previous compute) ----
      float4 ta[2];                             // A: 64x32 = 512 float4
#pragma unroll
      for (int it = 0; it < 2; ++it) {
        int idx = tid + it * 256;
        int r = idx >> 3, kp = (idx & 7) << 2;
        ta[it] = *(const float4*)&emb[(rowBase + r) * DDIM + kk + kp];
      }
      float2 tw[8];                             // W tile: 32(k) x 128(n), pairs along n
#pragma unroll
      for (int it = 0; it < 8; ++it) {
        int idx = tid + it * 256;
        int k = idx >> 6, np = (idx & 63) << 1;
        int c = n0 + np;
        tw[it] = make_float2(0.f, 0.f);
        if (c < CDIM) tw[it] = *(const float2*)&W[(kk + k) * CDIM + c];  // CDIM even: pair safe
      }
      __syncthreads();  // previous k-step finished reading LDS
      // ---- convert + store phase ----
#pragma unroll
      for (int it = 0; it < 2; ++it) {
        int idx = tid + it * 256;
        int r = idx >> 3, kp = (idx & 7) << 2;
        stage4(&Ah[r][kp], &Al[r][kp], ta[it]);
      }
#pragma unroll
      for (int it = 0; it < 8; ++it) {
        int idx = tid + it * 256;
        int k = idx >> 6, np = (idx & 63) << 1;
        __bf16 h, l;
        bsplit(tw[it].x, h, l); Bh[np][k] = h;     Bl[np][k] = l;
        bsplit(tw[it].y, h, l); Bh[np + 1][k] = h; Bl[np + 1][k] = l;
      }
      __syncthreads();
      // ---- compute phase ----
      v16bf ah0 = load_a16(&Ah[wm * 32 + ln][0], half);
      v16bf al0 = load_a16(&Al[wm * 32 + ln][0], half);
      v16bf ah1 = load_a16(&Ah[wm * 32 + 16 + ln][0], half);
      v16bf al1 = load_a16(&Al[wm * 32 + 16 + ln][0], half);
      v16bf bh0 = load_b16(&Bh[wn * 32 + ln][0], half);
      v16bf bl0 = load_b16(&Bl[wn * 32 + ln][0], half);
      v16bf bh1 = load_b16(&Bh[wn * 32 + 16 + ln][0], half);
      v16bf bl1 = load_b16(&Bl[wn * 32 + 16 + ln][0], half);

      acc00 = wmma3(ah0, al0, bh0, bl0, acc00);
      acc01 = wmma3(ah0, al0, bh1, bl1, acc01);
      acc10 = wmma3(ah1, al1, bh0, bl0, acc10);
      acc11 = wmma3(ah1, al1, bh1, bl1, acc11);
    }

    __syncthreads();
#pragma unroll
    for (int r = 0; r < 8; ++r) {
      int m0 = wm * 32 + half * 8 + r;
      Lt[m0][wn * 32 + ln]           = acc00[r];
      Lt[m0][wn * 32 + 16 + ln]      = acc01[r];
      Lt[m0 + 16][wn * 32 + ln]      = acc10[r];
      Lt[m0 + 16][wn * 32 + 16 + ln] = acc11[r];
    }
    __syncthreads();

    // epilogue: 2 threads per row (adjacent lanes), shuffle-merge partials
    if (tid < 128) {
      const int row = tid >> 1;
      const int qh = tid & 1;                 // column half [qh*64, qh*64+64)
      const int base = qh * 64;
      float lm = -3.0e38f, lsum = 0.f, lbest = -3.0e38f, lylog = -3.0e38f;
      int lbidx = 0;
      for (int j = 0; j < 64; ++j) {
        int c = n0 + base + j;
        if (c >= CDIM) break;
        float v = Lt[row][base + j];
        if (v > lm) { lsum = lsum * __expf(lm - v) + 1.f; lm = v; }
        else        { lsum += __expf(v - lm); }
        if (v > lbest) { lbest = v; lbidx = c; }
        if (c == yv) lylog = v;
      }
      float om    = __shfl_xor(lm, 1);
      float osum  = __shfl_xor(lsum, 1);
      float obest = __shfl_xor(lbest, 1);
      int   obidx = __shfl_xor(lbidx, 1);
      float oylog = __shfl_xor(lylog, 1);
      float tm   = fmaxf(lm, om);
      float tsum = lsum * __expf(lm - tm) + osum * __expf(om - tm);
      float blow  = qh ? obest : lbest;  int ilow  = qh ? obidx : lbidx;
      float bhigh = qh ? lbest : obest;  int ihigh = qh ? lbidx : obidx;
      float tbest; int tbidx;
      if (bhigh > blow) { tbest = bhigh; tbidx = ihigh; } else { tbest = blow; tbidx = ilow; }
      float tylog = fmaxf(lylog, oylog);
      float nm = fmaxf(rm, tm);
      rsum = rsum * __expf(rm - nm) + tsum * __expf(tm - nm);
      rm = nm;
      if (tbest > rbest) { rbest = tbest; rbidx = tbidx; }
      rylog = fmaxf(rylog, tylog);
    }
  }
  if (tid < 128 && (tid & 1) == 0) {
    int o = chunk * B_ROWS + rowBase + (tid >> 1);
    pm[o] = rm; ps[o] = rsum; pbest[o] = rbest; pylog[o] = rylog; pbidx[o] = rbidx;
  }
}

// ---------------- MMD: fused Gram GEMM + 7-sigma Gaussian sums -----------------
// grid = (32 jtiles, 64 itiles, 3 pairs); block = 256. Tile 64(i) x 128(j).
// Double-buffered LDS: stage k+1 while computing k; one barrier per k-step.
__global__ __launch_bounds__(256) void k_mmd(const float* __restrict__ emb,
                                             const int* __restrict__ main_src,
                                             const int* __restrict__ targ_src,
                                             const float* __restrict__ normM,
                                             const float* __restrict__ normT,
                                             float* __restrict__ partials) {
  __shared__ __bf16 Ah[2][64][ASTRIDE], Al[2][64][ASTRIDE];     // 20,480 B
  __shared__ __bf16 Bh[2][128][BSTRIDE], Bl[2][128][BSTRIDE];   // 40,960 B
  __shared__ float red[256];
  const int tid = threadIdx.x;
  const int jt = blockIdx.x, it = blockIdx.y, which = blockIdx.z;  // 0:MM 1:MT 2:TT
  const int* srcA = (which == 2) ? targ_src : main_src;
  const int* srcB = (which == 0) ? main_src : targ_src;
  const float* nA = (which == 2) ? normT : normM;
  const float* nB = (which == 0) ? normM : normT;
  const int i0 = it * 64, j0 = jt * 128;
  const int lane = tid & 31, wv = tid >> 5;
  const int wm = wv & 1, wn = wv >> 1;
  const int half = lane >> 4, ln = lane & 15;

  // ---- prologue: stage k=0 into buffer 0 ----
  {
    float4 ta[2], tb[4];
#pragma unroll
    for (int itr = 0; itr < 2; ++itr) {
      int idx = tid + itr * 256;
      int r = idx >> 3, kp = (idx & 7) << 2;
      int s = srcA[i0 + r];
      ta[itr] = make_float4(0.f, 0.f, 0.f, 0.f);
      if (s >= 0) ta[itr] = *(const float4*)&emb[s * DDIM + kp];
    }
#pragma unroll
    for (int itr = 0; itr < 4; ++itr) {
      int idx = tid + itr * 256;
      int r = idx >> 3, kp = (idx & 7) << 2;
      int s = srcB[j0 + r];
      tb[itr] = make_float4(0.f, 0.f, 0.f, 0.f);
      if (s >= 0) tb[itr] = *(const float4*)&emb[s * DDIM + kp];
    }
#pragma unroll
    for (int itr = 0; itr < 2; ++itr) {
      int idx = tid + itr * 256;
      int r = idx >> 3, kp = (idx & 7) << 2;
      stage4(&Ah[0][r][kp], &Al[0][r][kp], ta[itr]);
    }
#pragma unroll
    for (int itr = 0; itr < 4; ++itr) {
      int idx = tid + itr * 256;
      int r = idx >> 3, kp = (idx & 7) << 2;
      stage4(&Bh[0][r][kp], &Bl[0][r][kp], tb[itr]);
    }
  }
  __syncthreads();

  v8f z = {};
  v8f acc00 = z, acc01 = z, acc10 = z, acc11 = z;
  int p = 0;

  for (int kk = 0; kk < DDIM; kk += 32) {
    const int kn = kk + 32;
    const bool more = (kn < DDIM);
    // ---- issue next-tile global loads ----
    float4 ta[2], tb[4];
    if (more) {
#pragma unroll
      for (int itr = 0; itr < 2; ++itr) {
        int idx = tid + itr * 256;
        int r = idx >> 3, kp = (idx & 7) << 2;
        int s = srcA[i0 + r];
        ta[itr] = make_float4(0.f, 0.f, 0.f, 0.f);
        if (s >= 0) ta[itr] = *(const float4*)&emb[s * DDIM + kn + kp];
      }
#pragma unroll
      for (int itr = 0; itr < 4; ++itr) {
        int idx = tid + itr * 256;
        int r = idx >> 3, kp = (idx & 7) << 2;
        int s = srcB[j0 + r];
        tb[itr] = make_float4(0.f, 0.f, 0.f, 0.f);
        if (s >= 0) tb[itr] = *(const float4*)&emb[s * DDIM + kn + kp];
      }
    }
    // ---- compute from buffer p ----
    {
      v16bf ah0 = load_a16(&Ah[p][wm * 32 + ln][0], half);
      v16bf al0 = load_a16(&Al[p][wm * 32 + ln][0], half);
      v16bf ah1 = load_a16(&Ah[p][wm * 32 + 16 + ln][0], half);
      v16bf al1 = load_a16(&Al[p][wm * 32 + 16 + ln][0], half);
      v16bf bh0 = load_b16(&Bh[p][wn * 32 + ln][0], half);
      v16bf bl0 = load_b16(&Bl[p][wn * 32 + ln][0], half);
      v16bf bh1 = load_b16(&Bh[p][wn * 32 + 16 + ln][0], half);
      v16bf bl1 = load_b16(&Bl[p][wn * 32 + 16 + ln][0], half);

      acc00 = wmma3(ah0, al0, bh0, bl0, acc00);
      acc01 = wmma3(ah0, al0, bh1, bl1, acc01);
      acc10 = wmma3(ah1, al1, bh0, bl0, acc10);
      acc11 = wmma3(ah1, al1, bh1, bl1, acc11);
    }
    // ---- convert + store into buffer p^1 ----
    if (more) {
      const int q = p ^ 1;
#pragma unroll
      for (int itr = 0; itr < 2; ++itr) {
        int idx = tid + itr * 256;
        int r = idx >> 3, kp = (idx & 7) << 2;
        stage4(&Ah[q][r][kp], &Al[q][r][kp], ta[itr]);
      }
#pragma unroll
      for (int itr = 0; itr < 4; ++itr) {
        int idx = tid + itr * 256;
        int r = idx >> 3, kp = (idx & 7) << 2;
        stage4(&Bh[q][r][kp], &Bl[q][r][kp], tb[itr]);
      }
    }
    __syncthreads();
    p ^= 1;
  }

  // epilogue: d2 = |a|^2 + |b|^2 - 2 a.b ; 7-bandwidth Gaussian sum
  float s = 0.f;
  const int jA = j0 + wn * 32 + ln;
  const float dj0 = nB[jA], dj1 = nB[jA + 16];
#pragma unroll
  for (int r = 0; r < 8; ++r) {
    int ia = i0 + wm * 32 + half * 8 + r;
    float di0 = nA[ia], di1 = nA[ia + 16];
    s += expsum7(di0 + dj0 - 2.f * acc00[r]);
    s += expsum7(di0 + dj1 - 2.f * acc01[r]);
    s += expsum7(di1 + dj0 - 2.f * acc10[r]);
    s += expsum7(di1 + dj1 - 2.f * acc11[r]);
  }
  red[tid] = s;
  __syncthreads();
  for (int off = 128; off > 0; off >>= 1) {
    if (tid < off) red[tid] += red[tid + off];
    __syncthreads();
  }
  if (tid == 0) partials[which * 2048 + it * 32 + jt] = red[0];
}

// ---------------- final merge: CE chunks + MMD partials -> 4 scalars -----------
__global__ __launch_bounds__(256) void k_final(const float* pm, const float* ps,
                                               const float* pbest, const float* pylog,
                                               const int* pbidx, const int* __restrict__ y,
                                               const float* partials, float* out) {
  __shared__ float rA[256], rB[256], rC[256];
  const int tid = threadIdx.x;
  float lsum = 0.f, asum = 0.f;
  for (int row = tid; row < B_ROWS; row += 256) {
    float m = -3.0e38f;
    for (int c = 0; c < NCHUNKS; ++c) m = fmaxf(m, pm[c * B_ROWS + row]);
    float sum = 0.f, best = -3.0e38f, ylog = -3.0e38f;
    int bidx = 0;
    for (int c = 0; c < NCHUNKS; ++c) {
      int o = c * B_ROWS + row;
      sum += ps[o] * __expf(pm[o] - m);
      if (pbest[o] > best) { best = pbest[o]; bidx = pbidx[o]; }
      ylog = fmaxf(ylog, pylog[o]);
    }
    lsum += (m + logf(sum)) - ylog;
    asum += (bidx == y[row]) ? 1.f : 0.f;
  }
  float msum = 0.f;
  for (int i = tid; i < 3 * 2048; i += 256) {
    float w = (i >= 2048 && i < 4096) ? -2.f : 1.f;
    msum += w * partials[i];
  }
  rA[tid] = lsum; rB[tid] = asum; rC[tid] = msum;
  __syncthreads();
  for (int off = 128; off > 0; off >>= 1) {
    if (tid < off) { rA[tid] += rA[tid + off]; rB[tid] += rB[tid + off]; rC[tid] += rC[tid + off]; }
    __syncthreads();
  }
  if (tid == 0) {
    out[0] = rA[0] / (float)B_ROWS;
    float L = rC[0] / ((float)HALF_ROWS * (float)HALF_ROWS);
    out[1] = L;
    out[2] = rB[0] / (float)B_ROWS;
    out[3] = L;
  }
}

extern "C" void kernel_launch(void* const* d_in, const int* in_sizes, int n_in,
                              void* d_out, int out_size, void* d_ws, size_t ws_size,
                              hipStream_t stream) {
  const float* emb = (const float*)d_in[0];
  const float* W   = (const float*)d_in[1];
  const int*   y   = (const int*)d_in[2];
  const int*   y_d = (const int*)d_in[3];
  float* out = (float*)d_out;

  int*   main_src = (int*)d_ws;
  int*   targ_src = main_src + HALF_ROWS;
  float* normM    = (float*)(targ_src + HALF_ROWS);
  float* normT    = normM + HALF_ROWS;
  float* pm       = normT + HALF_ROWS;
  float* ps       = pm + NCHUNKS * B_ROWS;
  float* pbest    = ps + NCHUNKS * B_ROWS;
  float* pylog    = pbest + NCHUNKS * B_ROWS;
  int*   pbidx    = (int*)(pylog + NCHUNKS * B_ROWS);
  float* partials = (float*)(pbidx + NCHUNKS * B_ROWS);

  k_init_idx<<<dim3(16), dim3(256), 0, stream>>>(main_src, targ_src);
  k_scatter<<<dim3(32), dim3(256), 0, stream>>>(y_d, main_src, targ_src);
  k_norms<<<dim3(B_ROWS), dim3(128), 0, stream>>>(emb, main_src, targ_src, normM, normT);
  k_ce<<<dim3(NCHUNKS, B_ROWS / 64), dim3(256), 0, stream>>>(emb, W, y, pm, ps, pbest, pylog, pbidx);
  k_mmd<<<dim3(32, 64, 3), dim3(256), 0, stream>>>(emb, main_src, targ_src, normM, normT, partials);
  k_final<<<dim3(1), dim3(256), 0, stream>>>(pm, ps, pbest, pylog, pbidx, y, partials, out);
}
